// CodeBook_51573967290755
// MI455X (gfx1250) — matrix operand order repaired and checked by reading, not compiled, not run on hardware
//
#include <hip/hip_runtime.h>
#include <hip/hip_bf16.h>

// CDNA5 / gfx1250 WMMA types
typedef __attribute__((ext_vector_type(16))) __bf16 v16bf;
typedef __attribute__((ext_vector_type(8)))  float  v8f;
typedef __attribute__((ext_vector_type(4)))  float  f32x4;

constexpr int N_PTS = 131072;
constexpr int DIM   = 64;
constexpr int KCB   = 1024;

constexpr int ROWS_PER_BLOCK = 128;
constexpr int THREADS        = 512;           // 16 wave32s -> 4 waves/SIMD
constexpr int WAVES          = THREADS / 32;
constexpr int COL_TILES      = KCB / 16;      // 64 column tiles of 16
constexpr int TILES_PER_WAVE = COL_TILES / WAVES; // 4

// LDS layout (dynamic shared; 316,416 B <= 320 KB per WGP)
constexpr int SMEM_BYTES =
    KCB * DIM * 2              // CB_HI (bf16)
  + KCB * DIM * 2              // CB_LO (bf16)
  + ROWS_PER_BLOCK * DIM * 2   // X_HI
  + ROWS_PER_BLOCK * DIM * 2   // X_LO
  + KCB * 4                    // c_sq
  + ROWS_PER_BLOCK * 4         // x_sq
  + ROWS_PER_BLOCK * WAVES * 4 // redv
  + ROWS_PER_BLOCK * WAVES * 4 // redi
  + ROWS_PER_BLOCK * 4;        // codes

__device__ __forceinline__ unsigned short f2bf(float f) {
  unsigned int u = __float_as_uint(f);
  u += 0x7FFFu + ((u >> 16) & 1u);      // round-to-nearest-even
  return (unsigned short)(u >> 16);
}
__device__ __forceinline__ float bf2f(unsigned short h) {
  return __uint_as_float(((unsigned int)h) << 16);
}
// sum of squares of 8 reconstructed (hi+lo) bf16 pairs packed in two uint4s
__device__ __forceinline__ float sq8(uint4 h, uint4 l) {
  float s = 0.f;
  const unsigned hw[4] = {h.x, h.y, h.z, h.w};
  const unsigned lw[4] = {l.x, l.y, l.z, l.w};
  #pragma unroll
  for (int i = 0; i < 4; ++i) {
    float a = bf2f((unsigned short)(hw[i] & 0xFFFFu)) +
              bf2f((unsigned short)(lw[i] & 0xFFFFu));
    float b = bf2f((unsigned short)(hw[i] >> 16)) +
              bf2f((unsigned short)(lw[i] >> 16));
    s += a * a + b * b;
  }
  return s;
}

union V16 { v16bf v; uint4 q[2]; };

#define WMMA_BF16(A, B, C) \
  __builtin_amdgcn_wmma_f32_16x16x32_bf16(false, (A), false, (B), (short)0, (C), false, false)

__global__ __launch_bounds__(THREADS, 1)
void codebook_wmma_kernel(const float* __restrict__ x,
                          const float* __restrict__ cb,
                          float* __restrict__ out_cv,
                          int*   __restrict__ out_codes,
                          float* __restrict__ out_l2)
{
  extern __shared__ char smem[];
  unsigned short* CB_HI = (unsigned short*)smem;
  unsigned short* CB_LO = CB_HI + KCB * DIM;
  unsigned short* X_HI  = CB_LO + KCB * DIM;
  unsigned short* X_LO  = X_HI + ROWS_PER_BLOCK * DIM;
  float* c_sq      = (float*)(X_LO + ROWS_PER_BLOCK * DIM);
  float* x_sq      = c_sq + KCB;
  float* redv      = x_sq + ROWS_PER_BLOCK;
  int*   redi      = (int*)(redv + ROWS_PER_BLOCK * WAVES);
  int*   codes_lds = redi + ROWS_PER_BLOCK * WAVES;

  const int tid  = threadIdx.x;
  const int lane = tid & 31;
  const int wave = tid >> 5;
  const int half = lane >> 4;   // which 16-lane half of the wave
  const int ln   = lane & 15;
  const int rowBase = blockIdx.x * ROWS_PER_BLOCK;

  // ---- Stage 1: codebook -> LDS as bf16 hi/lo split (coalesced, keep L2-hot) ----
  {
    const float4* cb4 = (const float4*)cb;
    uint2* hi2 = (uint2*)CB_HI;
    uint2* lo2 = (uint2*)CB_LO;
    for (int f = tid; f < KCB * DIM / 4; f += THREADS) {
      float4 v = cb4[f];
      unsigned short h0 = f2bf(v.x), h1 = f2bf(v.y), h2 = f2bf(v.z), h3 = f2bf(v.w);
      unsigned short l0 = f2bf(v.x - bf2f(h0));
      unsigned short l1 = f2bf(v.y - bf2f(h1));
      unsigned short l2 = f2bf(v.z - bf2f(h2));
      unsigned short l3 = f2bf(v.w - bf2f(h3));
      hi2[f] = make_uint2((unsigned)h0 | ((unsigned)h1 << 16),
                          (unsigned)h2 | ((unsigned)h3 << 16));
      lo2[f] = make_uint2((unsigned)l0 | ((unsigned)l1 << 16),
                          (unsigned)l2 | ((unsigned)l3 << 16));
    }
  }
  // ---- Stage 2: x row tile -> LDS (non-temporal streaming reads) ----
  {
    const f32x4* x4 = (const f32x4*)(x + (size_t)rowBase * DIM);
    uint2* hi2 = (uint2*)X_HI;
    uint2* lo2 = (uint2*)X_LO;
    for (int f = tid; f < ROWS_PER_BLOCK * DIM / 4; f += THREADS) {
      f32x4 v = __builtin_nontemporal_load(x4 + f);
      unsigned short h0 = f2bf(v[0]), h1 = f2bf(v[1]), h2 = f2bf(v[2]), h3 = f2bf(v[3]);
      unsigned short l0 = f2bf(v[0] - bf2f(h0));
      unsigned short l1 = f2bf(v[1] - bf2f(h1));
      unsigned short l2 = f2bf(v[2] - bf2f(h2));
      unsigned short l3 = f2bf(v[3] - bf2f(h3));
      hi2[f] = make_uint2((unsigned)h0 | ((unsigned)h1 << 16),
                          (unsigned)h2 | ((unsigned)h3 << 16));
      lo2[f] = make_uint2((unsigned)l0 | ((unsigned)l1 << 16),
                          (unsigned)l2 | ((unsigned)l3 << 16));
    }
  }
  __syncthreads();

  // ---- Stage 3: squared norms (deterministic, vector LDS reads) ----
  for (int c = tid; c < KCB; c += THREADS) {
    const uint4* h4 = (const uint4*)(CB_HI + c * DIM);
    const uint4* l4 = (const uint4*)(CB_LO + c * DIM);
    float s = 0.f;
    #pragma unroll
    for (int i = 0; i < DIM / 8; ++i) s += sq8(h4[i], l4[i]);
    c_sq[c] = s;
  }
  if (tid < ROWS_PER_BLOCK) {
    const uint4* h4 = (const uint4*)(X_HI + tid * DIM);
    const uint4* l4 = (const uint4*)(X_LO + tid * DIM);
    float s = 0.f;
    #pragma unroll
    for (int i = 0; i < DIM / 8; ++i) s += sq8(h4[i], l4[i]);
    x_sq[tid] = s;
  }
  __syncthreads();

  // ---- Stage 4: WMMA GEMM tiles + fused distance + argmin ----
  const int colBase0 = wave * TILES_PER_WAVE * 16;

  for (int s = 0; s < ROWS_PER_BLOCK / 16; ++s) {
    // A operands (16x32 bf16 layout): lane row = s*16 + ln for both halves;
    // lanes 0-15 hold K 0..15 (+d0), lanes 16-31 hold K 16..31 (+d0).
    const unsigned short* abh = X_HI + (s * 16 + ln) * DIM;
    const unsigned short* abl = X_LO + (s * 16 + ln) * DIM;
    V16 ah0, ah1, al0, al1;
    ah0.q[0] = *(const uint4*)(abh + 0  + 16 * half);
    ah0.q[1] = *(const uint4*)(abh + 8  + 16 * half);
    ah1.q[0] = *(const uint4*)(abh + 32 + 16 * half);
    ah1.q[1] = *(const uint4*)(abh + 40 + 16 * half);
    al0.q[0] = *(const uint4*)(abl + 0  + 16 * half);
    al0.q[1] = *(const uint4*)(abl + 8  + 16 * half);
    al1.q[0] = *(const uint4*)(abl + 32 + 16 * half);
    al1.q[1] = *(const uint4*)(abl + 40 + 16 * half);

    float xs[8];
    #pragma unroll
    for (int j = 0; j < 8; ++j) xs[j] = x_sq[s * 16 + j + 8 * half];

    float bestv[8]; int besti[8];
    #pragma unroll
    for (int j = 0; j < 8; ++j) { bestv[j] = __builtin_inff(); besti[j] = 0; }

    // Fully unrolled so the scheduler can interleave the next tile's LDS
    // loads / WMMAs into this tile's hazard windows (no v_nop padding).
    #pragma unroll
    for (int t = 0; t < TILES_PER_WAVE; ++t) {
      const int colBase = colBase0 + t * 16;
      const int col = colBase + ln;
      // B operands (32x16 bf16 layout): lane col = colBase + ln;
      // lanes 0-15 hold K 0..15 (+d0), lanes 16-31 hold K 16..31 (+d0).
      const unsigned short* bbh = CB_HI + col * DIM + 16 * half;
      const unsigned short* bbl = CB_LO + col * DIM + 16 * half;
      V16 bh0, bh1, bl0, bl1;
      bh0.q[0] = *(const uint4*)(bbh + 0);  bh0.q[1] = *(const uint4*)(bbh + 8);
      bh1.q[0] = *(const uint4*)(bbh + 32); bh1.q[1] = *(const uint4*)(bbh + 40);
      bl0.q[0] = *(const uint4*)(bbl + 0);  bl0.q[1] = *(const uint4*)(bbl + 8);
      bl1.q[0] = *(const uint4*)(bbl + 32); bl1.q[1] = *(const uint4*)(bbl + 40);

      // bf16x3 on THREE independent accumulator chains (one per product term);
      // >=2 independent WMMAs separate any same-chain RAW pair.
      v8f acc_hh = {}, acc_hl = {}, acc_lh = {};
      acc_hh = WMMA_BF16(ah0.v, bh0.v, acc_hh);
      acc_hl = WMMA_BF16(ah0.v, bl0.v, acc_hl);
      acc_lh = WMMA_BF16(al0.v, bh0.v, acc_lh);
      acc_hh = WMMA_BF16(ah1.v, bh1.v, acc_hh);
      acc_hl = WMMA_BF16(ah1.v, bl1.v, acc_hl);
      acc_lh = WMMA_BF16(al1.v, bh1.v, acc_lh);

      const float csq = c_sq[col];
      #pragma unroll
      for (int j = 0; j < 8; ++j) {
        // C/D layout: VGPR j, lanes 0-15 -> row j, lanes 16-31 -> row j+8
        float dot = (acc_hh[j] + acc_hl[j]) + acc_lh[j];
        float v2 = xs[j] - 2.0f * dot + csq;
        size_t row = (size_t)(rowBase + s * 16 + j + 8 * half);
        __builtin_nontemporal_store(v2, &out_l2[row * KCB + col]); // streaming out
        if (v2 < bestv[j]) { bestv[j] = v2; besti[j] = col; }
      }
    }

    // reduce (val,idx) across the 16 lanes of each half, then publish per wave
    #pragma unroll
    for (int j = 0; j < 8; ++j) {
      float v = bestv[j]; int i = besti[j];
      #pragma unroll
      for (int off = 8; off >= 1; off >>= 1) {
        float ov = __shfl_xor(v, off, 32);
        int   oi = __shfl_xor(i, off, 32);
        if (ov < v || (ov == v && oi < i)) { v = ov; i = oi; }
      }
      if (ln == 0) {
        int rl = s * 16 + j + 8 * half;
        redv[rl * WAVES + wave] = v;
        redi[rl * WAVES + wave] = i;
      }
    }
  }
  __syncthreads();

  // ---- Stage 5: cross-wave argmin, emit codes ----
  if (tid < ROWS_PER_BLOCK) {
    float bv = redv[tid * WAVES]; int bi = redi[tid * WAVES];
    #pragma unroll
    for (int w = 1; w < WAVES; ++w) {
      float v = redv[tid * WAVES + w]; int i = redi[tid * WAVES + w];
      if (v < bv || (v == bv && i < bi)) { bv = v; bi = i; }
    }
    codes_lds[tid] = bi;
    out_codes[rowBase + tid] = bi;
  }
  __syncthreads();

  // ---- Stage 6: gather exact f32 code vectors from global codebook (L2-hot) ----
  for (int f = tid; f < ROWS_PER_BLOCK * DIM; f += THREADS) {
    int r = f >> 6, d = f & 63;
    int code = codes_lds[r];
    float v = cb[(size_t)code * DIM + d];
    __builtin_nontemporal_store(v, &out_cv[(size_t)(rowBase + r) * DIM + d]);
  }
}

extern "C" void kernel_launch(void* const* d_in, const int* in_sizes, int n_in,
                              void* d_out, int out_size, void* d_ws, size_t ws_size,
                              hipStream_t stream) {
  (void)in_sizes; (void)n_in; (void)out_size; (void)d_ws; (void)ws_size;
  const float* x  = (const float*)d_in[0];
  const float* cb = (const float*)d_in[1];

  float* outF      = (float*)d_out;
  float* out_cv    = outF;                                   // [N, 64] f32
  int*   out_codes = (int*)(outF + (size_t)N_PTS * DIM);     // [N] int32
  float* out_l2    = outF + (size_t)N_PTS * DIM + N_PTS;     // [N, K] f32

  // >64KB dynamic LDS: raise the limit (idempotent, capture-safe host call)
  static_assert(SMEM_BYTES <= 320 * 1024, "LDS budget");
  hipFuncSetAttribute((const void*)codebook_wmma_kernel,
                      hipFuncAttributeMaxDynamicSharedMemorySize, SMEM_BYTES);

  dim3 grid(N_PTS / ROWS_PER_BLOCK);  // 1024 blocks
  codebook_wmma_kernel<<<grid, THREADS, SMEM_BYTES, stream>>>(
      x, cb, out_cv, out_codes, out_l2);
}
